// Attention_40922448396911
// MI455X (gfx1250) — matrix operand order repaired and checked by reading, not compiled, not run on hardware
//
#include <hip/hip_runtime.h>
#include <hip/hip_bf16.h>
#include <stdint.h>

// ---------- types ----------
typedef __attribute__((ext_vector_type(16))) __bf16          v16bf;
typedef __attribute__((ext_vector_type(8)))  float           v8f;
typedef __attribute__((ext_vector_type(4)))  float           v4f;
typedef __attribute__((ext_vector_type(4)))  unsigned short  v4us;
typedef __attribute__((ext_vector_type(8)))  unsigned short  v8us;
typedef __attribute__((ext_vector_type(16))) unsigned short  v16us;
typedef __attribute__((ext_vector_type(4)))  unsigned int    v4u;
typedef __attribute__((ext_vector_type(4)))  int             v4i;
typedef __attribute__((ext_vector_type(8)))  int             v8i;

#if __has_builtin(__builtin_amdgcn_tensor_load_to_lds)
#define HAVE_TDM 1
#else
#define HAVE_TDM 0
#endif

__device__ __forceinline__ unsigned short f32_to_bf16(float f) {
    unsigned u = __float_as_uint(f);
    unsigned r = u + 0x7FFFu + ((u >> 16) & 1u);   // round-to-nearest-even
    return (unsigned short)(r >> 16);
}

__device__ __forceinline__ unsigned lds_offset_of(const void* p) {
    // generic pointer into LDS aperture: low 32 bits are the LDS byte offset
    return (unsigned)(unsigned long long)(uintptr_t)p;
}

// ---- TDM: 2D tile (tile_d0 contiguous elems x tile_d1 rows) -> padded LDS ----
// bf16 rows of 32 elems = 16 DWORDs; pad_interval=3 (16dw), pad_amount=3 (4dw)
// => LDS row pitch 80B = LDSS(40) halves.
__device__ __forceinline__ void tdm_load_2d(const unsigned short* g,
                                            unsigned lds_off,
                                            unsigned tile_d0, unsigned tile_d1,
                                            unsigned long long stride0)
{
#if HAVE_TDM
    unsigned long long ga = (unsigned long long)(uintptr_t)g;
    const unsigned td0 = 0x7FFFFFFFu, td1 = 0x7FFFFFFFu;  // generous (no OOB)
    v4u g0 = { 1u,                                   // count=1, user descriptor
               lds_off,                              // lds_addr
               (unsigned)ga,                         // global_addr[31:0]
               (unsigned)((ga >> 32) & 0x01FFFFFFu) | 0x80000000u }; // [56:32] | type=2
    v8i g1;
    g1[0] = (int)((1u << 16) | (1u << 20) | (3u << 22) | (3u << 25)); // 2B, pad en, 16dw/4dw
    g1[1] = (int)((td0 & 0xFFFFu) << 16);                            // tensor_dim0 lo
    g1[2] = (int)(((td0 >> 16) & 0xFFFFu) | ((td1 & 0xFFFFu) << 16)); // d0 hi | d1 lo
    g1[3] = (int)(((td1 >> 16) & 0xFFFFu) | ((tile_d0 & 0xFFFFu) << 16)); // d1 hi | tile0
    g1[4] = (int)(tile_d1 & 0xFFFFu);                                // tile1 | tile2=0
    g1[5] = (int)(unsigned)(stride0 & 0xFFFFFFFFu);                  // dim0 stride lo
    g1[6] = (int)(unsigned)((stride0 >> 32) & 0xFFFFu);              // stride hi | d1str=0
    g1[7] = 0;
    v4i z4 = {0, 0, 0, 0};
#if defined(__clang_major__) && (__clang_major__ >= 23)
    v8i z8 = {0, 0, 0, 0, 0, 0, 0, 0};
    __builtin_amdgcn_tensor_load_to_lds(g0, g1, z4, z4, z8, 0);
#else
    __builtin_amdgcn_tensor_load_to_lds(g0, g1, z4, z4, 0);
#endif
#else
    (void)g; (void)lds_off; (void)tile_d0; (void)tile_d1; (void)stride0;
#endif
}

__device__ __forceinline__ void tdm_wait() {
#if HAVE_TDM
    __builtin_amdgcn_s_wait_tensorcnt(0);
#endif
}

// ---------- fp32 -> bf16 conversion, 4-wide ----------
__global__ void __launch_bounds__(256)
cvt_f32_bf16_v4(const v4f* __restrict__ in, v4us* __restrict__ out, int n4) {
    int i = blockIdx.x * 256 + threadIdx.x;
    if (i < n4) {
        v4f v = in[i];
        v4us o;
#pragma unroll
        for (int j = 0; j < 4; ++j) o[j] = f32_to_bf16(v[j]);
        out[i] = o;
    }
}

// ---------- generic strided batched bf16 GEMM via WMMA + TDM staging ----------
// C[b](m,n) = sum_k A[b](m,k)*B[b](k,n) (+bias[n]), fp32 accumulate.
// AM: 0 => A K-contiguous (a_sk==1); 1 => A M-contiguous (a_sm==1)
// BM: 0 => B N-contiguous (b_sn==1); 1 => B K-contiguous (b_sk==1)
// TDMA/TDMB: stage A/B tiles with tensor_load_to_lds (requires K-contiguous op)
// Block: 256 threads = 8 wave32. Tile 128x128, K-step 32, double-buffered LDS.
#define LDSS 40   // padded LDS row stride (halves); rows stay 16B aligned

template<int AM, int BM, int TDMA, int TDMB>
__global__ void __launch_bounds__(256)
wmma_gemm_bf16(const unsigned short* __restrict__ A,
               const unsigned short* __restrict__ Bm,
               void* __restrict__ C,
               const float* __restrict__ bias,
               int K,
               long long a_sm, long long a_sk,
               long long b_sk, long long b_sn,
               long long a_batch, long long b_batch, long long c_batch,
               int ldc, int store_bf16)
{
    __shared__ alignas(16) unsigned short At[2][128 * LDSS]; // [m][k]
    __shared__ alignas(16) unsigned short Bt[2][128 * LDSS]; // [n][k] (K-major)

    const int bz = blockIdx.z;
    A  += (long long)bz * a_batch;
    Bm += (long long)bz * b_batch;

    const int tileM = blockIdx.y * 128;
    const int tileN = blockIdx.x * 128;

    const int t     = threadIdx.x;   // 0..255
    const int wave  = t >> 5;        // 0..7
    const int lane  = t & 31;
    const int waveM = wave >> 1;     // 0..3 -> 32-row strip
    const int waveN = wave & 1;      // 0..1 -> 64-col strip
    const int lm    = lane & 15;
    const int hi    = lane >> 4;

    // uniform tile bases for TDM (K-contiguous operand layouts)
    const unsigned short* aTile = A + (long long)tileM * a_sm;
    const unsigned short* bTile = Bm + (long long)tileN * b_sn;

    // ---- manual staging coords (2 x v8us per thread per operand) ----
    int am[2], ak[2], bn[2], bk[2];
    const unsigned short* ap[2] = {nullptr, nullptr};
    const unsigned short* bp[2] = {nullptr, nullptr};
#pragma unroll
    for (int i = 0; i < 2; ++i) {
        int idx = t + i * 256;                         // 0..511
        if (!TDMA) {
            if (AM == 0) { am[i] = idx >> 2;  ak[i] = (idx & 3)  * 8; }
            else         { ak[i] = idx >> 4;  am[i] = (idx & 15) * 8; }
            ap[i] = A + (long long)(tileM + am[i]) * a_sm + (long long)ak[i] * a_sk;
        }
        if (!TDMB) {
            if (BM == 0) { bk[i] = idx >> 4;  bn[i] = (idx & 15) * 8; }
            else         { bn[i] = idx >> 2;  bk[i] = (idx & 3)  * 8; }
            bp[i] = Bm + (long long)bk[i] * b_sk + (long long)(tileN + bn[i]) * b_sn;
        }
    }
    const long long aStep = 32 * a_sk;
    const long long bStep = 32 * b_sk;

    v8f acc[2][4];
    const v8f vz = {0.f,0.f,0.f,0.f,0.f,0.f,0.f,0.f};
#pragma unroll
    for (int mf = 0; mf < 2; ++mf)
#pragma unroll
        for (int nt = 0; nt < 4; ++nt) acc[mf][nt] = vz;

    v8us ra[2], rb[2];

    // ---- preload tile 0 ----
    if ((TDMA || TDMB) && t < 32) {
        if (TDMA) tdm_load_2d(aTile, lds_offset_of(&At[0][0]), 32, 128, (unsigned long long)a_sm);
        if (TDMB) tdm_load_2d(bTile, lds_offset_of(&Bt[0][0]), 32, 128, (unsigned long long)b_sn);
        tdm_wait();
    }
    if (!TDMA) {
#pragma unroll
        for (int i = 0; i < 2; ++i) { ra[i] = *(const v8us*)ap[i]; ap[i] += aStep; }
#pragma unroll
        for (int i = 0; i < 2; ++i) {
            if (AM == 0) *(v8us*)&At[0][am[i] * LDSS + ak[i]] = ra[i];
            else { for (int j = 0; j < 8; ++j) At[0][(am[i] + j) * LDSS + ak[i]] = ra[i][j]; }
        }
    }
    if (!TDMB) {
#pragma unroll
        for (int i = 0; i < 2; ++i) { rb[i] = *(const v8us*)bp[i]; bp[i] += bStep; }
#pragma unroll
        for (int i = 0; i < 2; ++i) {
            if (BM == 1) *(v8us*)&Bt[0][bn[i] * LDSS + bk[i]] = rb[i];
            else { for (int j = 0; j < 8; ++j) Bt[0][(bn[i] + j) * LDSS + bk[i]] = rb[i][j]; }
        }
    }
    __syncthreads();

    const int nsteps = K >> 5;
    for (int s = 0; s < nsteps; ++s) {
        const int buf = s & 1;
        const bool more = (s + 1 < nsteps);

        if (more) {     // launch next tile's transfers early (hide behind WMMA)
            if ((TDMA || TDMB) && t < 32) {
                const int nb = buf ^ 1;
                if (TDMA) tdm_load_2d(aTile + (long long)(s + 1) * 32,
                                      lds_offset_of(&At[nb][0]), 32, 128,
                                      (unsigned long long)a_sm);
                if (TDMB) tdm_load_2d(bTile + (long long)(s + 1) * 32,
                                      lds_offset_of(&Bt[nb][0]), 32, 128,
                                      (unsigned long long)b_sn);
            }
            if (!TDMA) {
#pragma unroll
                for (int i = 0; i < 2; ++i) { ra[i] = *(const v8us*)ap[i]; ap[i] += aStep; }
            }
            if (!TDMB) {
#pragma unroll
                for (int i = 0; i < 2; ++i) { rb[i] = *(const v8us*)bp[i]; bp[i] += bStep; }
            }
        }

        // ---- compute on current buffer: 8 x v_wmma ----
#pragma unroll
        for (int mf = 0; mf < 2; ++mf) {
            const unsigned short* arp =
                &At[buf][(waveM * 32 + mf * 16 + lm) * LDSS + hi * 8];
            v8us alo = *(const v8us*)(arp);
            v8us ahi = *(const v8us*)(arp + 16);
            v16bf af = __builtin_bit_cast(v16bf, __builtin_shufflevector(
                alo, ahi, 0,1,2,3,4,5,6,7,8,9,10,11,12,13,14,15));
#pragma unroll
            for (int nt = 0; nt < 4; ++nt) {
                const unsigned short* brp =
                    &Bt[buf][(waveN * 64 + nt * 16 + lm) * LDSS + hi * 8];
                v8us blo = *(const v8us*)(brp);
                v8us bhi = *(const v8us*)(brp + 16);
                v16bf bfv = __builtin_bit_cast(v16bf, __builtin_shufflevector(
                    blo, bhi, 0,1,2,3,4,5,6,7,8,9,10,11,12,13,14,15));
                acc[mf][nt] = __builtin_amdgcn_wmma_f32_16x16x32_bf16(
                    false, af, false, bfv, (short)0, acc[mf][nt], false, false);
            }
        }

        if (more) {     // commit next tile
            const int nb = buf ^ 1;
            if (!TDMA) {
#pragma unroll
                for (int i = 0; i < 2; ++i) {
                    if (AM == 0) *(v8us*)&At[nb][am[i] * LDSS + ak[i]] = ra[i];
                    else { for (int j = 0; j < 8; ++j) At[nb][(am[i] + j) * LDSS + ak[i]] = ra[i][j]; }
                }
            }
            if (!TDMB) {
#pragma unroll
                for (int i = 0; i < 2; ++i) {
                    if (BM == 1) *(v8us*)&Bt[nb][bn[i] * LDSS + bk[i]] = rb[i];
                    else { for (int j = 0; j < 8; ++j) Bt[nb][(bn[i] + j) * LDSS + bk[i]] = rb[i][j]; }
                }
            }
            if ((TDMA || TDMB) && t < 32) tdm_wait();   // TDM done before publish
        }
        __syncthreads();
    }

    // ---- epilogue ----
    const long long cbase = (long long)bz * c_batch;
#pragma unroll
    for (int mf = 0; mf < 2; ++mf) {
#pragma unroll
        for (int nt = 0; nt < 4; ++nt) {
            int col = tileN + waveN * 64 + nt * 16 + lm;
            float bv = bias ? bias[col] : 0.f;
#pragma unroll
            for (int r = 0; r < 8; ++r) {
                int row = tileM + waveM * 32 + mf * 16 + hi * 8 + r;
                float v = acc[mf][nt][r] + bv;
                long long idx = cbase + (long long)row * ldc + col;
                if (store_bf16) ((unsigned short*)C)[idx] = f32_to_bf16(v);
                else            ((float*)C)[idx] = v;
            }
        }
    }
}

// ---------- row softmax (fp32 in, bf16 out), scale folded in ----------
__global__ void __launch_bounds__(256)
softmax_rows(const float* __restrict__ dots, unsigned short* __restrict__ attn,
             int ncols, float scale)
{
    long long row = blockIdx.x;
    const float* in = dots + row * ncols;
    unsigned short* out = attn + row * ncols;
    __shared__ float red[256];
    int t = threadIdx.x;

    float lmax = -3.402823466e38f;
    for (int c = t; c < ncols; c += 256) lmax = fmaxf(lmax, in[c] * scale);
    red[t] = lmax; __syncthreads();
    for (int s = 128; s > 0; s >>= 1) {
        if (t < s) red[t] = fmaxf(red[t], red[t + s]);
        __syncthreads();
    }
    float rmax = red[0]; __syncthreads();

    float lsum = 0.f;
    for (int c = t; c < ncols; c += 256) lsum += __expf(in[c] * scale - rmax);
    red[t] = lsum; __syncthreads();
    for (int s = 128; s > 0; s >>= 1) {
        if (t < s) red[t] += red[t + s];
        __syncthreads();
    }
    float inv = 1.0f / red[0];

    for (int c = t; c < ncols; c += 256)
        out[c] = f32_to_bf16(__expf(in[c] * scale - rmax) * inv);
}

// ---------- launcher ----------
extern "C" void kernel_launch(void* const* d_in, const int* in_sizes, int n_in,
                              void* d_out, int out_size, void* d_ws, size_t ws_size,
                              hipStream_t stream) {
    (void)in_sizes; (void)n_in; (void)out_size; (void)ws_size;
    const float* x    = (const float*)d_in[0];   // (4,4096,1024)
    const float* Wqkv = (const float*)d_in[1];   // (1024,3072)
    const float* bqkv = (const float*)d_in[2];   // (3072)
    const float* Wout = (const float*)d_in[3];   // (1024,1024)
    const float* bout = (const float*)d_in[4];   // (1024)
    float* out = (float*)d_out;                  // (4,4096,1024) fp32

    const int Bn = 4, N = 4096, DIM = 1024, HD = 1024, THR = 3072;
    const long long MN = (long long)Bn * N;      // 16384

    char* ws = (char*)d_ws;
    unsigned short* xb    = (unsigned short*)(ws);                 //  32 MiB (reused as outb)
    unsigned short* wqkvb = (unsigned short*)(ws +  33554432ll);   //   6 MiB
    unsigned short* woutb = (unsigned short*)(ws +  39845888ll);   //   2 MiB
    unsigned short* qkvb  = (unsigned short*)(ws +  41943040ll);   //  96 MiB
    float*          dotsf = (float*)         (ws + 142606336ll);   //  16 MiB
    unsigned short* attnb = (unsigned short*)(ws + 159383552ll);   //   8 MiB
    unsigned short* outb  = xb;  // x dead after GEMM1 (stream-ordered reuse)

    // 1) convert operands to bf16 (4-wide)
    {
        int nx4 = (int)(MN * DIM) / 4;
        cvt_f32_bf16_v4<<<dim3((nx4 + 255) / 256), dim3(256), 0, stream>>>(
            (const v4f*)x, (v4us*)xb, nx4);
        int nw4 = DIM * THR / 4;
        cvt_f32_bf16_v4<<<dim3((nw4 + 255) / 256), dim3(256), 0, stream>>>(
            (const v4f*)Wqkv, (v4us*)wqkvb, nw4);
        int no4 = HD * DIM / 4;
        cvt_f32_bf16_v4<<<dim3((no4 + 255) / 256), dim3(256), 0, stream>>>(
            (const v4f*)Wout, (v4us*)woutb, no4);
    }

    // 2) qkv = x @ Wqkv + bqkv : (16384 x 3072), K=1024, bf16 out. A via TDM.
    wmma_gemm_bf16<0,0,HAVE_TDM,0>
        <<<dim3(THR / 128, (int)(MN / 128), 1), dim3(256), 0, stream>>>(
        xb, wqkvb, qkvb, bqkv, DIM,
        /*a_sm*/ DIM, /*a_sk*/ 1, /*b_sk*/ THR, /*b_sn*/ 1,
        0, 0, 0, /*ldc*/ THR, /*store_bf16*/ 1);

    // 3) dots[b] = Q^T K : (1024 x 1024), K=4096, fp32 out (manual staging)
    wmma_gemm_bf16<1,0,0,0><<<dim3(HD / 128, HD / 128, Bn), dim3(256), 0, stream>>>(
        qkvb, qkvb + 1024, dotsf, nullptr, N,
        /*a_sm*/ 1, /*a_sk*/ THR, /*b_sk*/ THR, /*b_sn*/ 1,
        (long long)N * THR, (long long)N * THR,
        (long long)HD * HD, /*ldc*/ HD, /*store_bf16*/ 0);

    // 4) softmax over rows (scale = 1/sqrt(64)), bf16 out
    softmax_rows<<<dim3(Bn * HD), dim3(256), 0, stream>>>(dotsf, attnb, HD, 0.125f);

    // 5) outm[b] = attn @ V^T : (1024 x 4096), K=1024, bf16 out. A and B via TDM.
    wmma_gemm_bf16<0,1,HAVE_TDM,HAVE_TDM>
        <<<dim3(N / 128, HD / 128, Bn), dim3(256), 0, stream>>>(
        attnb, qkvb + 2048, outb, nullptr, HD,
        /*a_sm*/ HD, /*a_sk*/ 1, /*b_sk*/ 1, /*b_sn*/ THR,
        (long long)HD * HD, (long long)N * THR,
        (long long)HD * N, /*ldc*/ N, /*store_bf16*/ 1);

    // 6) final = flat(outm) @ Wout + bout : (16384 x 1024), K=1024, fp32 out. A via TDM.
    //    (torch reshape (b,hd,n)->(b,n,hd) is a flat reinterpret => plain GEMM)
    wmma_gemm_bf16<0,0,HAVE_TDM,0>
        <<<dim3(DIM / 128, (int)(MN / 128), 1), dim3(256), 0, stream>>>(
        outb, woutb, out, bout, HD,
        /*a_sm*/ HD, /*a_sk*/ 1, /*b_sk*/ DIM, /*b_sn*/ 1,
        0, 0, 0, /*ldc*/ DIM, /*store_bf16*/ 0);
}
